// InfiniTransformer_57329223467205
// MI455X (gfx1250) — matrix operand order repaired and checked by reading, not compiled, not run on hardware
//
#include <hip/hip_runtime.h>
#include <hip/hip_bf16.h>
#include <math.h>

// Problem constants (match reference)
#define BB   2
#define SS   4096
#define DDIM 2048
#define HH   16
#define DKK  128
#define DVV  128
#define DHH  8192
#define SEGL 512
#define NSEGS 8

typedef __attribute__((ext_vector_type(16))) __bf16 bf16x16;
typedef __attribute__((ext_vector_type(8)))  __bf16 bf16x8;
typedef __attribute__((ext_vector_type(4)))  __bf16 bf16x4;
typedef __attribute__((ext_vector_type(8)))  float  f32x8;
typedef int v4i __attribute__((vector_size(16)));  // matches async builtin param

// ---- CDNA5 async global->LDS DMA (guarded; falls back to sync copies) ----
#if defined(__has_builtin)
#if __has_builtin(__builtin_amdgcn_global_load_async_to_lds_b128)
#define HAVE_ASYNC_LDS 1
#endif
#if __has_builtin(__builtin_amdgcn_s_wait_asynccnt)
#define HAVE_ASYNC_WAIT 1
#endif
#endif
#ifndef HAVE_ASYNC_LDS
#define HAVE_ASYNC_LDS 0
#endif
#ifndef HAVE_ASYNC_WAIT
#define HAVE_ASYNC_WAIT 0
#endif

template <int NWAIT>
__device__ __forceinline__ void wait_async() {
#if HAVE_ASYNC_LDS
#if HAVE_ASYNC_WAIT
  __builtin_amdgcn_s_wait_asynccnt(NWAIT);
#else
  asm volatile("s_wait_asynccnt %0" ::"i"(NWAIT) : "memory");
#endif
#endif
}

__device__ __forceinline__ f32x8 zero8() {
  f32x8 z = {0.f, 0.f, 0.f, 0.f, 0.f, 0.f, 0.f, 0.f};
  return z;
}

__device__ __forceinline__ f32x8 wmma_bf16(bf16x16 a, bf16x16 b, f32x8 c) {
  // (neg_a, A, neg_b, B, c_mod, C, reuse_a, reuse_b)
  return __builtin_amdgcn_wmma_f32_16x16x32_bf16(false, a, false, b, (short)0, c,
                                                 false, false);
}

// A fragment (16x32 bf16, M x K) from row-major LDS a[m][k] (ld elems).
// lane L: m = L%16, half = L/16; elems 0..15 = a[m][half*16 + 0..15]
__device__ __forceinline__ bf16x16 load_a_frag(const __bf16* base, int ld) {
  const int lane = threadIdx.x & 31;
  const __bf16* p = base + (lane & 15) * ld + (lane >> 4) * 16;
  bf16x8 lo = *(const bf16x8*)p;
  bf16x8 hi = *(const bf16x8*)(p + 8);
  bf16x16 r;
#pragma unroll
  for (int i = 0; i < 8; ++i) { r[i] = lo[i]; r[i + 8] = hi[i]; }
  return r;
}

// B fragment (32x16 bf16, K x N) from *transposed* LDS bt[n][k] (ld elems).
// lane L: n = L%16, half = L/16; elems = bt[n][half*16 + 0..15]
__device__ __forceinline__ bf16x16 load_b_frag(const __bf16* base, int ld) {
  const int lane = threadIdx.x & 31;
  const __bf16* p = base + (lane & 15) * ld + (lane >> 4) * 16;
  return *(const bf16x16*)p;
}

// ---------------------------------------------------------------------------
// LayerNorm (optional residual, optional extra bf16 copy of the output)
// ---------------------------------------------------------------------------
__global__ __launch_bounds__(256) void ln_kernel(const float* __restrict__ x,
                                                 const float* __restrict__ res,
                                                 const float* __restrict__ g,
                                                 const float* __restrict__ bta,
                                                 float* __restrict__ out,
                                                 __bf16* __restrict__ outbf) {
  __shared__ float red[256];
  __shared__ float red2[256];
  const size_t row = blockIdx.x;
  const float* xr = x + row * DDIM;
  const float* rr = res ? res + row * DDIM : nullptr;
  float s = 0.f, s2 = 0.f;
  for (int i = threadIdx.x; i < DDIM; i += 256) {
    float v = xr[i] + (rr ? rr[i] : 0.0f);
    s += v;
    s2 += v * v;
  }
  red[threadIdx.x] = s;
  red2[threadIdx.x] = s2;
  __syncthreads();
  for (int off = 128; off > 0; off >>= 1) {
    if (threadIdx.x < off) {
      red[threadIdx.x] += red[threadIdx.x + off];
      red2[threadIdx.x] += red2[threadIdx.x + off];
    }
    __syncthreads();
  }
  const float mean = red[0] * (1.0f / DDIM);
  const float var  = red2[0] * (1.0f / DDIM) - mean * mean;
  const float inv  = rsqrtf(var + 1e-5f);
  for (int i = threadIdx.x; i < DDIM; i += 256) {
    float v = xr[i] + (rr ? rr[i] : 0.0f);
    const float o = (v - mean) * inv * g[i] + bta[i];
    if (out) out[row * DDIM + i] = o;
    if (outbf) outbf[row * DDIM + i] = (__bf16)o;
  }
}

// ---------------------------------------------------------------------------
// Weight convert + transpose: WT[n][k] (bf16) = W[k][n] (f32). 32x32 tiles.
// ---------------------------------------------------------------------------
__global__ __launch_bounds__(256) void wt_kernel(const float* __restrict__ W,
                                                 __bf16* __restrict__ WT,
                                                 int K, int N) {
  __shared__ __bf16 tile[32][33];
  const int nb = blockIdx.x * 32;
  const int kb = blockIdx.y * 32;
  const int tx = threadIdx.x & 31;
  const int ty = threadIdx.x >> 5;  // 0..7
#pragma unroll
  for (int i = 0; i < 32; i += 8)
    tile[ty + i][tx] = (__bf16)W[(size_t)(kb + ty + i) * N + nb + tx];
  __syncthreads();
#pragma unroll
  for (int i = 0; i < 32; i += 8)
    WT[(size_t)(nb + ty + i) * K + kb + tx] = tile[tx][ty + i];
}

// ---------------------------------------------------------------------------
// Stage 128 rows x 32 bf16 (64B rows) from global (row stride = gstride elems)
// into LDS (ld = ldl elems). Async DMA if available: 2 b128 ops per thread.
// ---------------------------------------------------------------------------
__device__ __forceinline__ void copy_rows_32(const __bf16* __restrict__ g,
                                             size_t gstride,
                                             __bf16* __restrict__ lds, int ldl) {
  const int t = threadIdx.x;
  const int quad = t & 3;   // 4 x 16B per 64B row
  const int rb = t >> 2;    // 0..63
#pragma unroll
  for (int it = 0; it < 2; ++it) {
    const int r = rb + it * 64;
    const __bf16* gp = g + (size_t)r * gstride + quad * 8;
    __bf16* lp = lds + r * ldl + quad * 8;
#if HAVE_ASYNC_LDS
    __builtin_amdgcn_global_load_async_to_lds_b128((v4i*)(void*)gp,
                                                   (v4i*)(void*)lp, 0, 0);
#else
    __builtin_prefetch(gp + 64, 0, 1);
    *(bf16x8*)lp = *(const bf16x8*)gp;
#endif
  }
}

// ---------------------------------------------------------------------------
// bf16 WMMA GEMM: C[M,N] = act(A[M,K] @ W[K,N] + bias), W given pre-transposed
// as WT[N][K] bf16. Block tile 128x128, 8 waves, wave w owns row-tile w x all
// 8 col-tiles. K step 32. Double-buffered LDS with async global->LDS DMA.
// Each stage = 4 async ops/thread (2 for A tile + 2 for WT tile).
// ---------------------------------------------------------------------------
template <int ACT, int OUTBF16>
__global__ __launch_bounds__(256) void gemm_wmma(const __bf16* __restrict__ A,
                                                 const __bf16* __restrict__ WT,
                                                 const float* __restrict__ bias,
                                                 void* __restrict__ Cptr,
                                                 int M, int N, int K) {
  __shared__ __align__(32) __bf16 As[2][128 * 40];  // A tile [m][k], ld=40
  __shared__ __align__(32) __bf16 Ws[2][128 * 48];  // W^T tile [n][k], ld=48
  const int n0 = blockIdx.x * 128;
  const int m0 = blockIdx.y * 128;
  const int t = threadIdx.x;
  const int w = t >> 5;
  const int lane = t & 31;
  const int half = lane >> 4;
  const int ln16 = lane & 15;

  const __bf16* Ablk = A + (size_t)m0 * K;
  const __bf16* Wblk = WT + (size_t)n0 * K;

  f32x8 acc[8];
#pragma unroll
  for (int i = 0; i < 8; ++i) acc[i] = zero8();

  // prologue: stage k0 = 0 into buffer 0
  copy_rows_32(Ablk, K, As[0], 40);
  copy_rows_32(Wblk, K, Ws[0], 48);

  for (int k0 = 0; k0 < K; k0 += 32) {
    const int cur = (k0 >> 5) & 1;
    const bool more = (k0 + 32) < K;
    if (more) {  // issue next tile's DMA; overlaps with this tile's WMMAs
      copy_rows_32(Ablk + k0 + 32, K, As[cur ^ 1], 40);
      copy_rows_32(Wblk + k0 + 32, K, Ws[cur ^ 1], 48);
      wait_async<4>();  // previous stage retired; 4 just-issued in flight
    } else {
      wait_async<0>();
    }
    __syncthreads();
    const bf16x16 af = load_a_frag(As[cur] + w * 16 * 40, 40);
#pragma unroll
    for (int nt = 0; nt < 8; ++nt) {
      const bf16x16 bfr = load_b_frag(Ws[cur] + nt * 16 * 48, 48);
      acc[nt] = wmma_bf16(af, bfr, acc[nt]);
    }
    __syncthreads();  // all waves done reading before this buf is re-staged
  }

  // epilogue: C layout: lane elem j -> (M = j + 8*half, N = ln16)
  const int mbase = m0 + w * 16 + half * 8;
#pragma unroll
  for (int nt = 0; nt < 8; ++nt) {
    const int n = n0 + nt * 16 + ln16;
    const float bv = bias ? bias[n] : 0.0f;
#pragma unroll
    for (int j = 0; j < 8; ++j) {
      float vv = acc[nt][j] + bv;
      if (ACT == 1) vv = 0.5f * vv * (1.0f + erff(vv * 0.70710678118654752f));
      const size_t off = (size_t)(mbase + j) * N + n;
      if (OUTBF16)
        ((__bf16*)Cptr)[off] = (__bf16)vv;
      else
        ((float*)Cptr)[off] = vv;
    }
  }
}

// ---------------------------------------------------------------------------
// Compressive-memory attention. One block per (b,h); 8 waves; sequential over
// segments. mem (128x128) lives in WMMA f32 accumulators (8 frags/wave),
// mirrored to LDS as bf16 memT[dv][dk] for the retrieval GEMM. Local causal
// attention is flash-style per 128-row chunk with LDS softmax.
// q/k/v layout: [B, S, H, 128] bf16. att out: same layout, bf16.
// ---------------------------------------------------------------------------
__device__ __forceinline__ void stage_kv_chunk(const __bf16* __restrict__ k,
                                               const __bf16* __restrict__ v,
                                               int b, int h, int srow0,
                                               __bf16* kB, __bf16* skT,
                                               __bf16* vT) {
  const int t = threadIdx.x;
  const int c4 = (t & 31) * 4;
  const int rb = t >> 5;
#pragma unroll 4
  for (int it = 0; it < 16; ++it) {
    const int r = rb + it * 8;
    const size_t base = ((size_t)(b * SS + srow0 + r) * HH + h) * DKK + c4;
    const bf16x4 kv = *(const bf16x4*)(k + base);
    const bf16x4 vv = *(const bf16x4*)(v + base);
#pragma unroll
    for (int i = 0; i < 4; ++i) {
      const float ke = (float)kv[i];
      kB[r * 128 + c4 + i] = kv[i];                              // raw k [key][dk]
      const float sk = ke > 0.0f ? ke + 1.0f : __expf(ke);       // elu+1
      skT[(c4 + i) * 128 + r] = (__bf16)sk;                      // [dk][key]
      vT[(c4 + i) * 128 + r]  = vv[i];                           // [dv][key]
    }
  }
}

__global__ __launch_bounds__(256) void attn_kernel(const __bf16* __restrict__ q,
                                                   const __bf16* __restrict__ k,
                                                   const __bf16* __restrict__ v,
                                                   const float* __restrict__ betas,
                                                   __bf16* __restrict__ att) {
  __shared__ __align__(32) __bf16 memT[DVV * DKK];   // mem^T [dv][dk] bf16
  __shared__ float zs[DKK];
  __shared__ __align__(32) __bf16 qA[128 * 128];     // raw q chunk [row][dk]
  __shared__ __align__(32) __bf16 sqA[128 * 128];    // elu(q)+1    [row][dk]
  __shared__ __align__(32) __bf16 kB[128 * 128];     // raw k chunk [key][dk]
  __shared__ __align__(32) __bf16 skT[128 * 128];    // elu(k)+1    [dk][key]
  __shared__ __align__(32) __bf16 vT[128 * 128];     // v chunk     [dv][key]
  __shared__ float sc[128 * 128];                    // scores f32
  __shared__ __align__(32) __bf16 pA[128 * 128];     // probs bf16  [row][key]
  __shared__ float mrun[128], lrun[128], fsc[128], den[128];

  const int bh = blockIdx.x;
  const int b = bh >> 4;
  const int h = bh & 15;
  const int t = threadIdx.x;
  const int w = t >> 5;
  const int lane = t & 31;
  const int half = lane >> 4;
  const int ln16 = lane & 15;
  const float scale = 0.088388347648318447f;  // 1/sqrt(128)

  for (int i = t; i < DVV * DKK; i += 256) memT[i] = (__bf16)0.0f;
  if (t < DKK) zs[t] = 1.0f / (float)DKK;
  f32x8 macc[8];
#pragma unroll
  for (int i = 0; i < 8; ++i) macc[i] = zero8();
  __syncthreads();

  for (int seg = 0; seg < NSEGS; ++seg) {
    const int segs0 = seg * SEGL;

    // ================= Pass A: outputs for this segment =================
    for (int qc = 0; qc < 4; ++qc) {
      __syncthreads();
      {  // stage q chunk (raw + elu+1)
        const int c4 = (t & 31) * 4;
        const int rb = t >> 5;
#pragma unroll 4
        for (int it = 0; it < 16; ++it) {
          const int r = rb + it * 8;
          const size_t base =
              ((size_t)(b * SS + segs0 + qc * 128 + r) * HH + h) * DKK + c4;
          const bf16x4 qv = *(const bf16x4*)(q + base);
#pragma unroll
          for (int i = 0; i < 4; ++i) {
            const float e = (float)qv[i];
            qA[r * 128 + c4 + i] = qv[i];
            const float sq = e > 0.0f ? e + 1.0f : __expf(e);
            sqA[r * 128 + c4 + i] = (__bf16)sq;
          }
        }
      }
      __syncthreads();
      // denominator sq @ z (z before this segment's update) + state init
      if (t < 128) {
        float s = 0.f;
        for (int dk = 0; dk < 128; ++dk) s += (float)sqA[t * 128 + dk] * zs[dk];
        den[t] = s;
        mrun[t] = -3.0e38f;
        lrun[t] = 0.0f;
      }
      // att_mem = sq @ mem (mem before update): A=sqA, B^T = memT[dv][dk]
      f32x8 am[8], o[8];
#pragma unroll
      for (int i = 0; i < 8; ++i) { am[i] = zero8(); o[i] = zero8(); }
#pragma unroll
      for (int nt = 0; nt < 8; ++nt) {
#pragma unroll
        for (int kk = 0; kk < 4; ++kk) {
          const bf16x16 a = load_a_frag(sqA + w * 16 * 128 + kk * 32, 128);
          const bf16x16 bb = load_b_frag(memT + nt * 16 * 128 + kk * 32, 128);
          am[nt] = wmma_bf16(a, bb, am[nt]);
        }
      }
      // flash loop over key chunks (causal -> kc <= qc)
      for (int kc = 0; kc <= qc; ++kc) {
        __syncthreads();
        stage_kv_chunk(k, v, b, h, segs0 + kc * 128, kB, skT, vT);
        __syncthreads();
        // scores = q @ k^T : A=qA[row][dk], B^T = kB[key][dk]
#pragma unroll
        for (int nt = 0; nt < 8; ++nt) {
          f32x8 s8 = zero8();
#pragma unroll
          for (int kk = 0; kk < 4; ++kk) {
            const bf16x16 a = load_a_frag(qA + w * 16 * 128 + kk * 32, 128);
            const bf16x16 bb = load_b_frag(kB + nt * 16 * 128 + kk * 32, 128);
            s8 = wmma_bf16(a, bb, s8);
          }
          const int qrow_c = w * 16 + half * 8;  // + j
          const int keyc = nt * 16 + ln16;
#pragma unroll
          for (int j = 0; j < 8; ++j) {
            float sval = s8[j] * scale;
            if (kc == qc && keyc > qrow_c + j) sval = -3.0e38f;  // causal mask
            sc[(qrow_c + j) * 128 + keyc] = sval;
          }
        }
        __syncthreads();
        // online softmax update (one thread per query row)
        if (t < 128) {
          float mx = -3.0e38f;
          for (int j = 0; j < 128; ++j) mx = fmaxf(mx, sc[t * 128 + j]);
          const float mnew = fmaxf(mrun[t], mx);
          const float fs = __expf(mrun[t] - mnew);
          float l = lrun[t] * fs;
          for (int j = 0; j < 128; ++j) {
            const float e = __expf(sc[t * 128 + j] - mnew);
            pA[t * 128 + j] = (__bf16)e;
            l += e;
          }
          mrun[t] = mnew;
          lrun[t] = l;
          fsc[t] = fs;
        }
        __syncthreads();
        // rescale O, accumulate O += P @ v : A=pA, B^T = vT[dv][key]
        float fsv[8];
#pragma unroll
        for (int j = 0; j < 8; ++j) fsv[j] = fsc[w * 16 + half * 8 + j];
#pragma unroll
        for (int nt = 0; nt < 8; ++nt) {
#pragma unroll
          for (int j = 0; j < 8; ++j) o[nt][j] *= fsv[j];
#pragma unroll
          for (int kk = 0; kk < 4; ++kk) {
            const bf16x16 a = load_a_frag(pA + w * 16 * 128 + kk * 32, 128);
            const bf16x16 bb = load_b_frag(vT + nt * 16 * 128 + kk * 32, 128);
            o[nt] = wmma_bf16(a, bb, o[nt]);
          }
        }
      }  // kc
      __syncthreads();
      // epilogue: att = gate * att_mem/den + (1-gate) * att_dot
      float linv[8], dinv[8];
#pragma unroll
      for (int j = 0; j < 8; ++j) {
        const int r = w * 16 + half * 8 + j;
        linv[j] = 1.0f / lrun[r];
        dinv[j] = 1.0f / den[r];
      }
#pragma unroll
      for (int nt = 0; nt < 8; ++nt) {
        const int dv = nt * 16 + ln16;
        const float bet = betas[h * DVV + dv];
        const float g = 1.0f / (1.0f + __expf(-bet));
#pragma unroll
        for (int j = 0; j < 8; ++j) {
          const int srow = segs0 + qc * 128 + w * 16 + half * 8 + j;
          const float val =
              g * am[nt][j] * dinv[j] + (1.0f - g) * o[nt][j] * linv[j];
          att[((size_t)(b * SS + srow) * HH + h) * DVV + dv] = (__bf16)val;
        }
      }
    }  // qc

    // ========== Pass B: mem += sk^T @ v (full segment), z += sum(sk) ========
    for (int kc = 0; kc < 4; ++kc) {
      __syncthreads();
      stage_kv_chunk(k, v, b, h, segs0 + kc * 128, kB, skT, vT);
      __syncthreads();
      if (t < 128) {  // z[dk] += sum over keys of sk
        float s = 0.f;
        for (int j = 0; j < 128; ++j) s += (float)skT[t * 128 + j];
        zs[t] += s;
      }
#pragma unroll
      for (int nt = 0; nt < 8; ++nt) {
#pragma unroll
        for (int kk = 0; kk < 4; ++kk) {
          const bf16x16 a = load_a_frag(skT + w * 16 * 128 + kk * 32, 128);
          const bf16x16 bb = load_b_frag(vT + nt * 16 * 128 + kk * 32, 128);
          macc[nt] = wmma_bf16(a, bb, macc[nt]);
        }
      }
    }
    __syncthreads();
    // refresh memT (bf16 mirror of register-resident mem) for next segment
#pragma unroll
    for (int nt = 0; nt < 8; ++nt) {
      const int dv = nt * 16 + ln16;
#pragma unroll
      for (int j = 0; j < 8; ++j) {
        const int dk = w * 16 + half * 8 + j;
        memT[dv * 128 + dk] = (__bf16)macc[nt][j];
      }
    }
    __syncthreads();
  }  // seg
}

// ---------------------------------------------------------------------------
extern "C" void kernel_launch(void* const* d_in, const int* in_sizes, int n_in,
                              void* d_out, int out_size, void* d_ws,
                              size_t ws_size, hipStream_t stream) {
  (void)in_sizes; (void)n_in; (void)out_size; (void)ws_size;
  const float* x    = (const float*)d_in[0];
  const float* ln_g = (const float*)d_in[1];
  const float* ln_b = (const float*)d_in[2];
  const float* Wq   = (const float*)d_in[3];
  const float* bq   = (const float*)d_in[4];
  const float* Wk   = (const float*)d_in[5];
  const float* bk   = (const float*)d_in[6];
  const float* Wv   = (const float*)d_in[7];
  const float* bv   = (const float*)d_in[8];
  const float* Wo   = (const float*)d_in[9];
  const float* bo   = (const float*)d_in[10];
  const float* betas= (const float*)d_in[11];
  const float* W1   = (const float*)d_in[12];
  const float* b1   = (const float*)d_in[13];
  const float* W2   = (const float*)d_in[14];
  const float* b2   = (const float*)d_in[15];
  float* out = (float*)d_out;

  char* ws = (char*)d_ws;
  const size_t MB = 1ull << 20;
  // Workspace plan (352 MB total with slot reuse):
  float*  xn   = (float*)(ws + 0 * MB);      // 64 MB f32 (live to end)
  __bf16* xnbf = (__bf16*)(ws + 64 * MB);    // 32 MB
  __bf16* WqT  = (__bf16*)(ws + 96 * MB);    // 8 MB each
  __bf16* WkT  = (__bf16*)(ws + 104 * MB);
  __bf16* WvT  = (__bf16*)(ws + 112 * MB);
  __bf16* WoT  = (__bf16*)(ws + 120 * MB);
  __bf16* W1T  = (__bf16*)(ws + 128 * MB);   // 32 MB ([8192][2048])
  __bf16* W2T  = (__bf16*)(ws + 160 * MB);   // 32 MB ([2048][8192])
  __bf16* qbf  = (__bf16*)(ws + 192 * MB);   // 32 MB each
  __bf16* kbf  = (__bf16*)(ws + 224 * MB);
  __bf16* vbf  = (__bf16*)(ws + 256 * MB);
  __bf16* attbf= (__bf16*)(ws + 288 * MB);   // 32 MB
  __bf16* abf  = (__bf16*)(ws + 320 * MB);   // 32 MB
  __bf16* hmid = (__bf16*)(ws + 192 * MB);   // 128 MB, reuses q/k/v/att slots
  float*  hb   = (float*)(ws + 96 * MB);     // 64 MB, reuses WqT..W1T slots

  const int M = BB * SS;  // 8192

  // 0. weights -> bf16, transposed (WT[n][k])
  wt_kernel<<<dim3(DDIM / 32, DDIM / 32), 256, 0, stream>>>(Wq, WqT, DDIM, DDIM);
  wt_kernel<<<dim3(DDIM / 32, DDIM / 32), 256, 0, stream>>>(Wk, WkT, DDIM, DDIM);
  wt_kernel<<<dim3(DDIM / 32, DDIM / 32), 256, 0, stream>>>(Wv, WvT, DDIM, DDIM);
  wt_kernel<<<dim3(DDIM / 32, DDIM / 32), 256, 0, stream>>>(Wo, WoT, DDIM, DDIM);
  wt_kernel<<<dim3(DHH / 32, DDIM / 32), 256, 0, stream>>>(W1, W1T, DDIM, DHH);
  wt_kernel<<<dim3(DDIM / 32, DHH / 32), 256, 0, stream>>>(W2, W2T, DHH, DDIM);

  // 1. xn = LN(x)  (f32 for residual + bf16 for GEMMs)
  ln_kernel<<<M, 256, 0, stream>>>(x, nullptr, ln_g, ln_b, xn, xnbf);

  // 2. Q/K/V projections (async-staged bf16 WMMA, bf16 out)
  dim3 gQKV(DDIM / 128, M / 128);
  gemm_wmma<0, 1><<<gQKV, 256, 0, stream>>>(xnbf, WqT, bq, qbf, M, DDIM, DDIM);
  gemm_wmma<0, 1><<<gQKV, 256, 0, stream>>>(xnbf, WkT, bk, kbf, M, DDIM, DDIM);
  gemm_wmma<0, 1><<<gQKV, 256, 0, stream>>>(xnbf, WvT, bv, vbf, M, DDIM, DDIM);

  // 3. compressive-memory attention (bf16 in/out)
  attn_kernel<<<BB * HH, 256, 0, stream>>>(qbf, kbf, vbf, betas, attbf);

  // 4. output projection: a = att @ Wo + bo (bf16 out)
  gemm_wmma<0, 1><<<gQKV, 256, 0, stream>>>(attbf, WoT, bo, abf, M, DDIM, DDIM);

  // 5. FFN: hmid = gelu(a @ W1 + b1) (bf16), h = hmid @ W2 + b2 (f32)
  dim3 gF1(DHH / 128, M / 128);
  gemm_wmma<1, 1><<<gF1, 256, 0, stream>>>(abf, W1T, b1, hmid, M, DHH, DDIM);
  dim3 gF2(DDIM / 128, M / 128);
  gemm_wmma<0, 0><<<gF2, 256, 0, stream>>>(hmid, W2T, b2, hb, M, DDIM, DHH);

  // 6. out = LN(h + xn)
  ln_kernel<<<M, 256, 0, stream>>>(hb, xn, ln_g, ln_b, out, nullptr);
}
